// FastMultiHeadAttention_9844065042579
// MI455X (gfx1250) — matrix operand order repaired and checked by reading, not compiled, not run on hardware
//
#include <hip/hip_runtime.h>

#define HIDDEN   1024
#define HEADS    16
#define HEAD_DIM 64
#define BATCH    2
#define SEQ      2048
#define MTOT     (BATCH * SEQ)   // 4096 tokens

typedef __attribute__((ext_vector_type(16))) _Float16 v16h;
typedef __attribute__((ext_vector_type(8)))  _Float16 v8h;
typedef __attribute__((ext_vector_type(8)))  float    v8f;

static __device__ __forceinline__ int lane_id() { return threadIdx.x & 31; }

static __device__ __forceinline__ v8f vzero8() {
  v8f z = {0.f, 0.f, 0.f, 0.f, 0.f, 0.f, 0.f, 0.f};
  return z;
}

// D = A*B + C, 16x16x32 f16 -> f32
static __device__ __forceinline__ v8f wmma_f16(v16h a, v16h b, v8f c) {
  return __builtin_amdgcn_wmma_f32_16x16x32_f16(false, a, false, b,
                                                (short)0, c, false, false);
}

// A-fragment (16x32, MxK): lane l -> row l&15; contiguous 16 K at (l>>4)*16.
static __device__ __forceinline__ v16h load_a_frag(const _Float16* p, int ld) {
  const int l = lane_id();
  return *(const v16h*)(p + (size_t)(l & 15) * ld + ((l >> 4) * 16));
}

// B-fragment (32x16, KxN): lane l -> col l&15, contiguous 16 K at (l>>4)*16.
// p points at (n0, k0) of a [N, K] row-major f16 matrix (torch W[n][k] layout).
static __device__ __forceinline__ v16h load_b_frag(const _Float16* p, int ld) {
  const int l = lane_id();
  return *(const v16h*)(p + (size_t)(l & 15) * ld + ((l >> 4) * 16));
}

// CDNA5 async global->LDS DMA (ASYNCcnt-tracked).  lds_off is the LDS byte
// offset (low 32 bits of the generic pointer per the LDS aperture mapping).
static __device__ __forceinline__ void async_load_b128(unsigned lds_off,
                                                       const void* gptr) {
  unsigned long long ga = (unsigned long long)(uintptr_t)gptr;
  asm volatile("global_load_async_to_lds_b128 %0, %1, off"
               :: "v"(lds_off), "v"(ga)
               : "memory");
}
static __device__ __forceinline__ void wait_async_le8() {
  asm volatile("s_wait_asynccnt 8" ::: "memory");
}
static __device__ __forceinline__ void wait_async_0() {
  asm volatile("s_wait_asynccnt 0" ::: "memory");
}

// ---------------------------------------------------------------------------
// f32 -> f16 pre-convert (vectorized, grid-stride).  n8 = elements/8.
// ---------------------------------------------------------------------------
__global__ void __launch_bounds__(256) cvt_f32_f16_kernel(
    const float* __restrict__ in, _Float16* __restrict__ out, int n8) {
  int i = blockIdx.x * blockDim.x + threadIdx.x;
  const int stride = gridDim.x * blockDim.x;
  for (; i < n8; i += stride) {
    v8f a = *(const v8f*)(in + (size_t)i * 8);
    *(v8h*)(out + (size_t)i * 8) = __builtin_convertvector(a, v8h);
  }
}

// ---------------------------------------------------------------------------
// f16 GEMM: C[M,N] = A[M,K] * Bw[N,K]^T + bias, 128x128 block, BK=64,
// double-buffered async global->LDS pipeline.  8 waves, each 64x32.
// MODE 0: QKV split epilogue (f16 Q scaled, K, V-transposed).
// MODE 1: plain f32 output epilogue.
// ---------------------------------------------------------------------------
template <int MODE>
__global__ void __launch_bounds__(256) gemm_f16_kernel(
    const _Float16* __restrict__ A, const _Float16* __restrict__ Bw,
    const float* __restrict__ bias,
    _Float16* __restrict__ Qd, _Float16* __restrict__ Kd,
    _Float16* __restrict__ Vtd, float* __restrict__ out) {
  constexpr int BK = 64;
  __shared__ alignas(32) _Float16 At[2][128 * BK];
  __shared__ alignas(32) _Float16 Bt[2][128 * BK];
  const int K = HIDDEN;
  const int n0 = blockIdx.x * 128;
  const int m0 = blockIdx.y * 128;
  const int tid = threadIdx.x;
  const int lane = tid & 31;
  const int wid = tid >> 5;
  const int wm = (wid >> 2) * 64;   // wave M offset: 0 / 64
  const int wn = (wid & 3) * 32;    // wave N offset: 0..96
  const int lrow = tid >> 1;        // 0..127 (tile fill row)
  const int lcol = (tid & 1) * 32;  // 0 / 32 halfs (64B per thread per tile)

  const _Float16* ga = A + (size_t)(m0 + lrow) * K + lcol;
  const _Float16* gb = Bw + (size_t)(n0 + lrow) * K + lcol;
  unsigned la[2], lb[2];
#pragma unroll
  for (int p = 0; p < 2; ++p) {
    la[p] = (unsigned)(uintptr_t)(&At[p][0] + lrow * BK + lcol);
    lb[p] = (unsigned)(uintptr_t)(&Bt[p][0] + lrow * BK + lcol);
  }
  auto issue = [&](int p, int k0) {
#pragma unroll
    for (int q = 0; q < 4; ++q) {
      async_load_b128(la[p] + q * 16, ga + k0 + q * 8);
      async_load_b128(lb[p] + q * 16, gb + k0 + q * 8);
    }
  };

  v8f acc[4][2];
#pragma unroll
  for (int i = 0; i < 4; ++i)
#pragma unroll
    for (int j = 0; j < 2; ++j) acc[i][j] = vzero8();

  issue(0, 0);
  for (int k0 = 0, it = 0; k0 < K; k0 += BK, ++it) {
    const int cur = it & 1;
    if (k0 + BK < K) {
      issue(cur ^ 1, k0 + BK);  // prefetch next tiles into the other buffer
      wait_async_le8();         // this buffer's 8 loads (issued last iter) done
    } else {
      wait_async_0();
    }
    __syncthreads();
    const _Float16* Ab = &At[cur][0];
    const _Float16* Bb = &Bt[cur][0];
#pragma unroll
    for (int kk = 0; kk < BK; kk += 32) {
      v16h bf0 = load_b_frag(Bb + (wn + 0) * BK + kk, BK);
      v16h bf1 = load_b_frag(Bb + (wn + 16) * BK + kk, BK);
#pragma unroll
      for (int i = 0; i < 4; ++i) {
        v16h af = load_a_frag(Ab + (wm + 16 * i) * BK + kk, BK);
        acc[i][0] = wmma_f16(af, bf0, acc[i][0]);
        acc[i][1] = wmma_f16(af, bf1, acc[i][1]);
      }
    }
    __syncthreads();  // all waves done reading before buffer is refilled
  }

  if constexpr (MODE == 0) {
    const int seg = n0 >> 10;  // 0=Q 1=K 2=V — block-uniform, branch ONCE
    if (seg == 0) {
#pragma unroll
      for (int j = 0; j < 2; ++j) {
        const int gn = n0 + wn + 16 * j + (lane & 15);
        const int h = (gn & (HIDDEN - 1)) >> 6, d = gn & 63;
        const float bv = bias[gn];
#pragma unroll
        for (int i = 0; i < 4; ++i)
#pragma unroll
          for (int e = 0; e < 8; ++e) {
            const int gm = m0 + wm + 16 * i + e + 8 * (lane >> 4);
            const size_t bh = (size_t)(gm >> 11) * HEADS + h;
            Qd[(bh * SEQ + (gm & (SEQ - 1))) * HEAD_DIM + d] =
                (_Float16)((acc[i][j][e] + bv) * 0.125f);
          }
      }
    } else if (seg == 1) {
#pragma unroll
      for (int j = 0; j < 2; ++j) {
        const int gn = n0 + wn + 16 * j + (lane & 15);
        const int h = (gn & (HIDDEN - 1)) >> 6, d = gn & 63;
        const float bv = bias[gn];
#pragma unroll
        for (int i = 0; i < 4; ++i)
#pragma unroll
          for (int e = 0; e < 8; ++e) {
            const int gm = m0 + wm + 16 * i + e + 8 * (lane >> 4);
            const size_t bh = (size_t)(gm >> 11) * HEADS + h;
            Kd[(bh * SEQ + (gm & (SEQ - 1))) * HEAD_DIM + d] =
                (_Float16)(acc[i][j][e] + bv);
          }
      }
    } else {
#pragma unroll
      for (int j = 0; j < 2; ++j) {
        const int gn = n0 + wn + 16 * j + (lane & 15);
        const int h = (gn & (HIDDEN - 1)) >> 6, d = gn & 63;
        const float bv = bias[gn];
#pragma unroll
        for (int i = 0; i < 4; ++i)
#pragma unroll
          for (int e = 0; e < 8; ++e) {
            const int gm = m0 + wm + 16 * i + e + 8 * (lane >> 4);
            const size_t bh = (size_t)(gm >> 11) * HEADS + h;
            Vtd[(bh * HEAD_DIM + d) * SEQ + (gm & (SEQ - 1))] =
                (_Float16)(acc[i][j][e] + bv);
          }
      }
    }
  } else {
#pragma unroll
    for (int j = 0; j < 2; ++j) {
      const int gn = n0 + wn + 16 * j + (lane & 15);
      const float bv = bias[gn];
#pragma unroll
      for (int i = 0; i < 4; ++i)
#pragma unroll
        for (int e = 0; e < 8; ++e) {
          const int gm = m0 + wm + 16 * i + e + 8 * (lane >> 4);
          out[(size_t)gm * HIDDEN + gn] = acc[i][j][e] + bv;
        }
    }
  }
}

// ---------------------------------------------------------------------------
// Flash-style attention, one wave per 16 query rows, 32 keys per block:
// 4 WMMAs QK^T -> mask -> online softmax -> LDS P re-shape -> 4 WMMAs P*V.
// ---------------------------------------------------------------------------
__global__ void __launch_bounds__(128) attn_kernel(
    const _Float16* __restrict__ Qd, const _Float16* __restrict__ Kd,
    const _Float16* __restrict__ Vtd, const int* __restrict__ mask,
    _Float16* __restrict__ ctx) {
  __shared__ alignas(32) _Float16 Pst[4][16 * 32];  // per-wave staging
  const int wid = threadIdx.x >> 5;
  const int lane = threadIdx.x & 31;
  const int bh = blockIdx.y;
  const int b = bh >> 4;
  const int h = bh & (HEADS - 1);
  const int q0 = blockIdx.x * 64 + wid * 16;

  const _Float16* Qp = Qd + (size_t)bh * SEQ * HEAD_DIM;
  const _Float16* Kp = Kd + (size_t)bh * SEQ * HEAD_DIM;
  const _Float16* Vp = Vtd + (size_t)bh * HEAD_DIM * SEQ;
  const int* mb = mask + (size_t)b * SEQ * SEQ;

  const v16h qf0 = load_a_frag(Qp + (size_t)q0 * HEAD_DIM + 0, HEAD_DIM);
  const v16h qf1 = load_a_frag(Qp + (size_t)q0 * HEAD_DIM + 32, HEAD_DIM);

  v8f acc[4];
#pragma unroll
  for (int t = 0; t < 4; ++t) acc[t] = vzero8();
  float mrun[8], lrun[8];
#pragma unroll
  for (int e = 0; e < 8; ++e) { mrun[e] = -3.0e38f; lrun[e] = 0.f; }

  const int c = lane & 15;
  const int hi = lane >> 4;
  _Float16* Ps = &Pst[wid][0];

  for (int kb = 0; kb < SEQ; kb += 32) {
    v8f s0 = vzero8(), s1 = vzero8();
    s0 = wmma_f16(qf0, load_b_frag(Kp + (size_t)kb * HEAD_DIM + 0, HEAD_DIM), s0);
    s0 = wmma_f16(qf1, load_b_frag(Kp + (size_t)kb * HEAD_DIM + 32, HEAD_DIM), s0);
    s1 = wmma_f16(qf0, load_b_frag(Kp + (size_t)(kb + 16) * HEAD_DIM + 0, HEAD_DIM), s1);
    s1 = wmma_f16(qf1, load_b_frag(Kp + (size_t)(kb + 16) * HEAD_DIM + 32, HEAD_DIM), s1);

#pragma unroll
    for (int e = 0; e < 8; ++e) {
      const int qrow = q0 + e + 8 * hi;
      if (mb[(size_t)qrow * SEQ + kb + c] == 0)      s0[e] = -1.0e9f;
      if (mb[(size_t)qrow * SEQ + kb + 16 + c] == 0) s1[e] = -1.0e9f;
    }

    float scale[8];
#pragma unroll
    for (int e = 0; e < 8; ++e) {
      float bm = fmaxf(s0[e], s1[e]);
#pragma unroll
      for (int off = 1; off < 16; off <<= 1)
        bm = fmaxf(bm, __shfl_xor(bm, off, 32));
      const float mnew = fmaxf(mrun[e], bm);
      scale[e] = __expf(mrun[e] - mnew);
      mrun[e] = mnew;
      s0[e] = __expf(s0[e] - mnew);
      s1[e] = __expf(s1[e] - mnew);
      float ps = s0[e] + s1[e];
#pragma unroll
      for (int off = 1; off < 16; off <<= 1) ps += __shfl_xor(ps, off, 32);
      lrun[e] = lrun[e] * scale[e] + ps;
    }
#pragma unroll
    for (int t = 0; t < 4; ++t)
#pragma unroll
      for (int e = 0; e < 8; ++e) acc[t][e] *= scale[e];

    // C-layout -> A-fragment layout via per-wave LDS staging (same-wave DS
    // ops are in order; compiler inserts s_wait_dscnt for the reload).
#pragma unroll
    for (int e = 0; e < 8; ++e) {
      Ps[(e + 8 * hi) * 32 + c] = (_Float16)s0[e];
      Ps[(e + 8 * hi) * 32 + 16 + c] = (_Float16)s1[e];
    }
    asm volatile("" ::: "memory");
    const v16h pf = load_a_frag(Ps, 32);
    asm volatile("" ::: "memory");
#pragma unroll
    for (int t = 0; t < 4; ++t)
      acc[t] = wmma_f16(pf, load_b_frag(Vp + (size_t)(t * 16) * SEQ + kb, SEQ),
                        acc[t]);
  }

#pragma unroll
  for (int t = 0; t < 4; ++t) {
#pragma unroll
    for (int e = 0; e < 8; ++e) {
      const int qrow = q0 + e + 8 * hi;
      const float inv = 1.0f / lrun[e];
      ctx[((size_t)b * SEQ + qrow) * HIDDEN + h * HEAD_DIM + t * 16 + c] =
          (_Float16)(acc[t][e] * inv);
    }
  }
}

extern "C" void kernel_launch(void* const* d_in, const int* in_sizes, int n_in,
                              void* d_out, int out_size, void* d_ws,
                              size_t ws_size, hipStream_t stream) {
  (void)in_sizes; (void)n_in; (void)out_size;
  const float* x     = (const float*)d_in[0];
  const int*   mask  = (const int*)d_in[1];
  const float* w_qkv = (const float*)d_in[2];
  const float* b_qkv = (const float*)d_in[3];
  const float* w_out = (const float*)d_in[4];
  const float* b_out = (const float*)d_in[5];
  float* out = (float*)d_out;

  const size_t nX  = (size_t)MTOT * HIDDEN;       // 4M
  const size_t nWq = (size_t)3 * HIDDEN * HIDDEN; // 3M
  const size_t nWo = (size_t)HIDDEN * HIDDEN;     // 1M
  const size_t per = (size_t)BATCH * HEADS * SEQ * HEAD_DIM;  // 4M

  _Float16* ws = (_Float16*)d_ws;
  _Float16* x_h   = ws;             // [MTOT, HIDDEN]
  _Float16* wq_h  = x_h + nX;       // [3H, H]
  _Float16* wo_h  = wq_h + nWq;     // [H, H]
  _Float16* Qd    = wo_h + nWo;     // [B,H,S,D] (pre-scaled 1/sqrt(D))
  _Float16* Kd    = Qd + per;       // [B,H,S,D]
  _Float16* Vtd   = Kd + per;       // [B,H,D,S]
  _Float16* ctx   = Vtd + per;      // [B,S,HIDDEN]
  const size_t need = (nX + nWq + nWo + 4 * per) * sizeof(_Float16);  // 48 MB
  if (ws_size < need) return;

  cvt_f32_f16_kernel<<<2048, 256, 0, stream>>>(x, x_h, (int)(nX / 8));
  cvt_f32_f16_kernel<<<2048, 256, 0, stream>>>(w_qkv, wq_h, (int)(nWq / 8));
  cvt_f32_f16_kernel<<<1024, 256, 0, stream>>>(w_out, wo_h, (int)(nWo / 8));

  gemm_f16_kernel<0><<<dim3(3 * HIDDEN / 128, MTOT / 128), 256, 0, stream>>>(
      x_h, wq_h, b_qkv, Qd, Kd, Vtd, nullptr);
  attn_kernel<<<dim3(SEQ / 64, BATCH * HEADS), 128, 0, stream>>>(
      Qd, Kd, Vtd, mask, ctx);
  gemm_f16_kernel<1><<<dim3(HIDDEN / 128, MTOT / 128), 256, 0, stream>>>(
      ctx, wo_h, b_out, nullptr, nullptr, nullptr, out);
}
